// StatefulAttention_67946382623249
// MI455X (gfx1250) — matrix operand order repaired and checked by reading, not compiled, not run on hardware
//
#include <hip/hip_runtime.h>
#include <hip/hip_bf16.h>

typedef __attribute__((ext_vector_type(16))) __bf16 v16bf;
typedef __attribute__((ext_vector_type(8)))  __bf16 v8bf;
typedef __attribute__((ext_vector_type(8)))  float  v8f;

namespace {
constexpr int B_  = 2, S_ = 2048, D_ = 1024, H_ = 16, HD_ = 64, MEM_ = 512;
constexpr int L_  = MEM_ + S_;   // 2560 keys after concat
constexpr int M_  = B_ * S_;     // 4096 rows for the dense GEMMs
constexpr float SCALE_ = 0.125f; // 1/sqrt(HD)
constexpr int   NBIN_  = 256;    // histogram bins for the 10th-percentile
constexpr float HLO_   = -16.f;  // scaled scores ~ N(0,1): [-16,16) covers them
constexpr float HSC_   = 8.f;    // 1 / bin width (bin width 0.125)
constexpr unsigned QCOUNT_ = 256;// 0.1 * 2560
}

__device__ __forceinline__ v8f v8f_zero() {
  v8f z;
#pragma unroll
  for (int i = 0; i < 8; ++i) z[i] = 0.f;
  return z;
}

__device__ __forceinline__ v16bf mk16(v8bf lo, v8bf hi) {
  v16bf r;
#pragma unroll
  for (int i = 0; i < 8; ++i) { r[i] = lo[i]; r[i + 8] = hi[i]; }
  return r;
}

// A-operand fragment gather: lane holds contiguous
// K = {2*koff..2*koff+15} of row (lane&15); koff = 0 / 8.
__device__ __forceinline__ v16bf load_afrag(const __bf16* rowp, int koff) {
  return mk16(*(const v8bf*)(rowp + 2 * koff), *(const v8bf*)(rowp + 2 * koff + 8));
}

__device__ __forceinline__ v8f wmma_bf16(v16bf a, v16bf b, v8f c) {
  // D = A(16x32 bf16) * B(32x16 bf16) + C(16x16 f32)
  return __builtin_amdgcn_wmma_f32_16x16x32_bf16(false, a, false, b,
                                                 (short)0, c, false, false);
}

__device__ __forceinline__ void atomicMaxF(float* addr, float v) {
  int* ai = (int*)addr;
  int cur = __float_as_int(*addr);
  while (__int_as_float(cur) < v) {
    int prev = atomicCAS(ai, cur, __float_as_int(v));
    if (prev == cur) break;
    cur = prev;
  }
}

// ---------------------------------------------------------------- conversions
__global__ void conv_f32_bf16(const float* __restrict__ src,
                              __bf16* __restrict__ dst, int n) {
  for (int i = blockIdx.x * blockDim.x + threadIdx.x; i < n;
       i += gridDim.x * blockDim.x)
    dst[i] = (__bf16)src[i];
}

// weights converted bf16 AND transposed: Wt[n*D + k] = W[k*D + n].
// B-operand WMMA fragments then become single contiguous 32B loads.
__global__ void conv_weights_t(const float* __restrict__ wq,
                               const float* __restrict__ wk,
                               const float* __restrict__ wv,
                               const float* __restrict__ wo,
                               __bf16* __restrict__ wallt, int n) {
  for (int i = blockIdx.x * blockDim.x + threadIdx.x; i < n;
       i += gridDim.x * blockDim.x) {
    int sel = i >> 20, off = i & ((1 << 20) - 1);
    int nn = off >> 10, kk = off & (D_ - 1);
    const float* s = (sel == 0) ? wq : (sel == 1) ? wk : (sel == 2) ? wv : wo;
    wallt[i] = (__bf16)s[(size_t)kk * D_ + nn];
  }
}

// splice past K (row-major keys) and past V (transposed, HD x L) bf16 buffers
__global__ void conv_past(const float* __restrict__ pk,
                          const float* __restrict__ pv,
                          __bf16* __restrict__ kcb,
                          __bf16* __restrict__ vcbt, int n) {
  for (int i = blockIdx.x * blockDim.x + threadIdx.x; i < n;
       i += gridDim.x * blockDim.x) {
    int hd = i & (HD_ - 1);
    int r  = i >> 6;
    int key = r % MEM_;
    int bh  = r / MEM_;
    kcb [((size_t)bh * L_  + key) * HD_ + hd] = (__bf16)pk[i];
    vcbt[((size_t)bh * HD_ + hd)  * L_ + key] = (__bf16)pv[i];
  }
}

// ------------------------------------------------------------- QKV projection
// C = X(4096x1024) @ W(1024x1024) + bias ; z = 0/1/2 -> Q/K/V
// LDS-free: A frags direct from row-major X, B frags direct from W^T.
// Block 128 = 4 waves; wave tile 32x64 (8 accumulators, 8 WMMA / k-step).
__global__ __launch_bounds__(128)
void gemm_qkv(const __bf16* __restrict__ Xb, const __bf16* __restrict__ Wallt,
              const float* __restrict__ bq, const float* __restrict__ bk,
              const float* __restrict__ bv,
              __bf16* __restrict__ Qb, __bf16* __restrict__ Kcb,
              __bf16* __restrict__ Vcbt,
              float* __restrict__ newk, float* __restrict__ newv) {
  const int z    = blockIdx.z;
  const int bm   = blockIdx.y * 128;
  const int bn   = blockIdx.x * 64;
  const int lane = threadIdx.x & 31, wave = threadIdx.x >> 5;
  const __bf16* Wt = Wallt + (size_t)z * (D_ * D_);

  const int r15  = lane & 15;
  const int koff = (lane < 16) ? 0 : 8;   // A fragment K sub-offset
  const int kk16 = (lane < 16) ? 0 : 16;  // B fragment K sub-offset
  const int wm   = bm + wave * 32;

  v8f acc[2][4];
#pragma unroll
  for (int mi = 0; mi < 2; ++mi)
#pragma unroll
    for (int j = 0; j < 4; ++j) acc[mi][j] = v8f_zero();

  const __bf16* a0 = Xb + (size_t)(wm + r15) * D_;        // rows wm..wm+15
  const __bf16* a1 = a0 + (size_t)16 * D_;                // rows wm+16..wm+31

  for (int k0 = 0; k0 < D_; k0 += 32) {
    v16bf af0 = load_afrag(a0 + k0, koff);
    v16bf af1 = load_afrag(a1 + k0, koff);
#pragma unroll
    for (int j = 0; j < 4; ++j) {
      // lane col n = bn + j*16 + (lane&15); contiguous K run from W^T
      const v16bf bfr =
          *(const v16bf*)(Wt + (size_t)(bn + j * 16 + r15) * D_ + k0 + kk16);
      acc[0][j] = wmma_bf16(af0, bfr, acc[0][j]);
      acc[1][j] = wmma_bf16(af1, bfr, acc[1][j]);
    }
  }

  const float* bias = (z == 0) ? bq : (z == 1) ? bk : bv;
  const int rowg = (lane < 16) ? 0 : 8;
#pragma unroll
  for (int j = 0; j < 4; ++j) {
    int n  = bn + j * 16 + r15;
    float bi = bias[n];
    int h  = n >> 6, hd = n & 63;
#pragma unroll
    for (int mi = 0; mi < 2; ++mi) {
#pragma unroll
      for (int r = 0; r < 8; ++r) {
        int m   = wm + mi * 16 + r + rowg;
        float v = acc[mi][j][r] + bi;
        int b = m >> 11, s = m & (S_ - 1);
        int bh = b * H_ + h;
        if (z == 0) {
          // pre-scale Q by 1/sqrt(HD): scores come out already scaled
          Qb[((size_t)bh * S_ + s) * HD_ + hd] = (__bf16)(v * SCALE_);
        } else if (z == 1) {
          Kcb[((size_t)bh * L_ + (MEM_ + s)) * HD_ + hd] = (__bf16)v;
          if (s >= S_ - MEM_)
            newk[((size_t)bh * MEM_ + (s - (S_ - MEM_))) * HD_ + hd] = v;
        } else {
          Vcbt[((size_t)bh * HD_ + hd) * L_ + (MEM_ + s)] = (__bf16)v;
          if (s >= S_ - MEM_)
            newv[((size_t)bh * MEM_ + (s - (S_ - MEM_))) * HD_ + hd] = v;
        }
      }
    }
  }
}

// ------------------------------------------------------- fused attention core
// One block = one (b, h, 16-row q tile). 4 waves split the 2560 keys.
// Pass 1: WMMA scores -> per-row histogram + masked row max.
// Pass 2: recompute scores, threshold + softmax numerator, WMMA P@V^T.
__global__ __launch_bounds__(128)
void attn_fused(const __bf16* __restrict__ Qb, const __bf16* __restrict__ Kcb,
                const __bf16* __restrict__ Vcbt, const int* __restrict__ amask,
                __bf16* __restrict__ Ctxb) {
  const int qt = blockIdx.x, h = blockIdx.y, b = blockIdx.z;
  const int tid = threadIdx.x, lane = tid & 31, wave = tid >> 5;
  const int bh  = b * H_ + h;
  const size_t qbase  = ((size_t)bh * S_ + qt * 16) * HD_;
  const size_t kbase  = (size_t)bh * L_ * HD_;
  const size_t vtbase = (size_t)bh * HD_ * L_;

  __shared__ unsigned s_hist[16][NBIN_];                 // 16 KB
  __shared__ float s_rowmax[16], s_thr[16], s_rsum[16];
  __shared__ __align__(32) __bf16 s_p[4][16][32];        // 4 KB per-wave P tile
  __shared__ float s_o[16][64];                          // 4 KB

  for (int i = tid; i < 16 * NBIN_; i += 128) (&s_hist[0][0])[i] = 0u;
  for (int i = tid; i < 16 * 64; i += 128) (&s_o[0][0])[i] = 0.f;
  if (tid < 16) { s_rowmax[tid] = -1e30f; s_rsum[tid] = 0.f; }
  __syncthreads();

  const int r15  = lane & 15;
  const int koff = (lane < 16) ? 0 : 8;
  const int kk16 = (lane < 16) ? 0 : 16;
  const int rowg = (lane < 16) ? 0 : 8;

  // Q fragments (16x64 bf16, K split 0..31 / 32..63), loaded once
  const __bf16* qp = Qb + qbase + (size_t)r15 * HD_;
  v16bf qf0 = load_afrag(qp, koff);
  v16bf qf1 = load_afrag(qp + 32, koff);

  // ---------------- pass 1: histogram + row max ----------------
  float lmax[8];
#pragma unroll
  for (int r = 0; r < 8; ++r) lmax[r] = -1e30f;

  for (int t = wave; t < L_ / 16; t += 4) {
    int key = t * 16 + r15;  // B-frag lane column == C-frag lane column
    const __bf16* kp = Kcb + kbase + (size_t)key * HD_;
    __builtin_prefetch(kp + 64 * HD_, 0, 1);   // global_prefetch next chunk
    v16bf kf0 = *(const v16bf*)(kp + kk16);
    v16bf kf1 = *(const v16bf*)(kp + 32 + kk16);
    v8f c = v8f_zero();
    c = wmma_bf16(qf0, kf0, c);
    c = wmma_bf16(qf1, kf1, c);
    int m = (key < MEM_) ? 1 : amask[b * S_ + key - MEM_];
#pragma unroll
    for (int r = 0; r < 8; ++r) {
      float sv = c[r];
      int row  = r + rowg;
      int bin  = (int)((sv - HLO_) * HSC_);
      bin = bin < 0 ? 0 : (bin > NBIN_ - 1 ? NBIN_ - 1 : bin);
      atomicAdd(&s_hist[row][bin], 1u);     // quantile is pre-mask (matches ref)
      if (m) lmax[r] = fmaxf(lmax[r], sv);
    }
  }
#pragma unroll
  for (int r = 0; r < 8; ++r) atomicMaxF(&s_rowmax[r + rowg], lmax[r]);
  __syncthreads();

  if (tid < 16) { // CDF scan -> 10th percentile bin lower edge
    unsigned cum = 0; int bsel = NBIN_ - 1;
    for (int i = 0; i < NBIN_; ++i) {
      cum += s_hist[tid][i];
      if (cum >= QCOUNT_) { bsel = i; break; }
    }
    s_thr[tid] = HLO_ + (float)bsel * (1.f / HSC_);
  }
  __syncthreads();

  // -------- pass 2: threshold + softmax numerator + P@V^T --------
  v8f o[4];
#pragma unroll
  for (int j = 0; j < 4; ++j) o[j] = v8f_zero();
  float lsum[8];
#pragma unroll
  for (int r = 0; r < 8; ++r) lsum[r] = 0.f;

  for (int tt = wave; tt < L_ / 32; tt += 4) {
#pragma unroll
    for (int hh = 0; hh < 2; ++hh) {
      int key = tt * 32 + hh * 16 + r15;
      const __bf16* kp = Kcb + kbase + (size_t)key * HD_;
      __builtin_prefetch(kp + 128 * HD_, 0, 1);
      v16bf kf0 = *(const v16bf*)(kp + kk16);
      v16bf kf1 = *(const v16bf*)(kp + 32 + kk16);
      v8f c = v8f_zero();
      c = wmma_bf16(qf0, kf0, c);
      c = wmma_bf16(qf1, kf1, c);
      int m = (key < MEM_) ? 1 : amask[b * S_ + key - MEM_];
#pragma unroll
      for (int r = 0; r < 8; ++r) {
        int row  = r + rowg;
        float sv = c[r];
        float p  = 0.f;
        if (m && sv >= s_thr[row]) {
          p = __expf(sv - s_rowmax[row]);
          lsum[r] += p;
        }
        s_p[wave][row][hh * 16 + r15] = (__bf16)p;
      }
    }
    // cross-lane DS dependency inside the wave: drain DS before reading back
    asm volatile("s_wait_dscnt 0" ::: "memory");

    const __bf16* pp = &s_p[wave][r15][kk16];
    v16bf pf = *(const v16bf*)pp;
#pragma unroll
    for (int j = 0; j < 4; ++j) {
      // V^T: lane col = hd (j*16 + r15), contiguous keys tt*32+kk16..+16
      const v16bf vf = *(const v16bf*)(Vcbt + vtbase +
                                       (size_t)(j * 16 + r15) * L_ +
                                       tt * 32 + kk16);
      o[j] = wmma_bf16(pf, vf, o[j]);
    }
  }

#pragma unroll
  for (int r = 0; r < 8; ++r) atomicAdd(&s_rsum[r + rowg], lsum[r]);
#pragma unroll
  for (int j = 0; j < 4; ++j)
#pragma unroll
    for (int r = 0; r < 8; ++r)
      atomicAdd(&s_o[r + rowg][j * 16 + r15], o[j][r]);
  __syncthreads();

  // normalize and emit bf16 context in (B,S,D) layout for the output GEMM
  for (int i = tid; i < 16 * 64; i += 128) {
    int row = i >> 6, col = i & 63;
    float inv = 1.f / s_rsum[row];
    int sg = qt * 16 + row;
    Ctxb[((size_t)(b * S_ + sg)) * D_ + h * HD_ + col] = (__bf16)(s_o[row][col] * inv);
  }
}

// ------------------------------------------------------------ output proj
__global__ __launch_bounds__(128)
void gemm_out(const __bf16* __restrict__ Ab, const __bf16* __restrict__ Wot,
              const float* __restrict__ bo, float* __restrict__ out) {
  const int bm   = blockIdx.y * 128;
  const int bn   = blockIdx.x * 64;
  const int lane = threadIdx.x & 31, wave = threadIdx.x >> 5;

  const int r15  = lane & 15;
  const int koff = (lane < 16) ? 0 : 8;
  const int kk16 = (lane < 16) ? 0 : 16;
  const int wm   = bm + wave * 32;

  v8f acc[2][4];
#pragma unroll
  for (int mi = 0; mi < 2; ++mi)
#pragma unroll
    for (int j = 0; j < 4; ++j) acc[mi][j] = v8f_zero();

  const __bf16* a0 = Ab + (size_t)(wm + r15) * D_;
  const __bf16* a1 = a0 + (size_t)16 * D_;

  for (int k0 = 0; k0 < D_; k0 += 32) {
    v16bf af0 = load_afrag(a0 + k0, koff);
    v16bf af1 = load_afrag(a1 + k0, koff);
#pragma unroll
    for (int j = 0; j < 4; ++j) {
      const v16bf bfr =
          *(const v16bf*)(Wot + (size_t)(bn + j * 16 + r15) * D_ + k0 + kk16);
      acc[0][j] = wmma_bf16(af0, bfr, acc[0][j]);
      acc[1][j] = wmma_bf16(af1, bfr, acc[1][j]);
    }
  }

  const int rowg = (lane < 16) ? 0 : 8;
#pragma unroll
  for (int j = 0; j < 4; ++j) {
    int n = bn + j * 16 + r15;
    float bi = bo[n];
#pragma unroll
    for (int mi = 0; mi < 2; ++mi)
#pragma unroll
      for (int r = 0; r < 8; ++r) {
        int m = wm + mi * 16 + r + rowg;
        out[(size_t)m * D_ + n] = acc[mi][j][r] + bi;
      }
  }
}

// ------------------------------------------------------------------ launcher
extern "C" void kernel_launch(void* const* d_in, const int* in_sizes, int n_in,
                              void* d_out, int out_size, void* d_ws, size_t ws_size,
                              hipStream_t stream) {
  const float* hidden = (const float*)d_in[0];
  const int*   amask  = (const int*)d_in[1];
  const float* pastk  = (const float*)d_in[2];
  const float* pastv  = (const float*)d_in[3];
  const float* Wq = (const float*)d_in[4];  const float* bq = (const float*)d_in[5];
  const float* Wk = (const float*)d_in[6];  const float* bk = (const float*)d_in[7];
  const float* Wv = (const float*)d_in[8];  const float* bv = (const float*)d_in[9];
  const float* Wo = (const float*)d_in[10]; const float* bo = (const float*)d_in[11];

  // workspace layout (bytes); total 54,525,952 B (~52 MB)
  uint8_t* w = (uint8_t*)d_ws;
  __bf16* Xb    = (__bf16*)(w + 0);          // X bf16           (M, D)
  __bf16* Wallt = (__bf16*)(w + 8388608);    // 4x W^T bf16      (N, K) each
  __bf16* Qb    = (__bf16*)(w + 16777216);   // Q bf16 scaled    (B,H,S,HD)
  __bf16* Kcb   = (__bf16*)(w + 25165824);   // K concat bf16    (B,H,L,HD)
  __bf16* Vcbt  = (__bf16*)(w + 35651584);   // V concat bf16 ^T (B,H,HD,L)
  __bf16* Ctxb  = (__bf16*)(w + 46137344);   // context bf16     (B,S,D)

  float* out  = (float*)d_out;               // (B,S,D)
  float* newk = out + (size_t)M_ * D_;       // (B,H,MEM,HD)
  float* newv = newk + (size_t)B_ * H_ * MEM_ * HD_;

  conv_f32_bf16 <<<2048, 256, 0, stream>>>(hidden, Xb, M_ * D_);
  conv_weights_t<<<2048, 256, 0, stream>>>(Wq, Wk, Wv, Wo, Wallt, 4 * D_ * D_);
  conv_past     <<<1024, 256, 0, stream>>>(pastk, pastv, Kcb, Vcbt,
                                           B_ * H_ * MEM_ * HD_);

  gemm_qkv<<<dim3(D_ / 64, M_ / 128, 3), 128, 0, stream>>>(
      Xb, Wallt, bq, bk, bv, Qb, Kcb, Vcbt, newk, newv);

  attn_fused<<<dim3(S_ / 16, H_, B_), 128, 0, stream>>>(
      Qb, Kcb, Vcbt, amask, Ctxb);

  gemm_out<<<dim3(D_ / 64, M_ / 128), 128, 0, stream>>>(
      Ctxb, Wallt + 3 * (size_t)(D_ * D_), bo, out);
}
